// GCN_29334626631943
// MI455X (gfx1250) — matrix operand order repaired and checked by reading, not compiled, not run on hardware
//
#include <hip/hip_runtime.h>
#include <hip/hip_bf16.h>

typedef __attribute__((ext_vector_type(16))) _Float16 v16h;
typedef __attribute__((ext_vector_type(8)))  float    v8f;

#define IN_FEATS 256
#define H_FEATS  256
#define OUT_FEATS 16
#define NUM_GRAPHS 64

__device__ __forceinline__ void atomAddF(float* p, float v) {
  unsafeAtomicAdd(p, v);   // lowers to global_atomic_add_f32 on gfx1250
}

// ---------------------------------------------------------------------------
// Degree kernels: deg starts at 1.0 (self loop), +1 per incoming edge,
// then deg := rsqrt(deg)  (deg >= 1 always, matches reference's where())
// ---------------------------------------------------------------------------
__global__ void k_deg_init(float* __restrict__ deg, int n) {
  int i = blockIdx.x * 256 + threadIdx.x;
  if (i < n) deg[i] = 1.0f;
}

__global__ void k_deg_edges(const int* __restrict__ dst, float* __restrict__ deg, int e) {
  int i = blockIdx.x * 256 + threadIdx.x;
  if (i < e) atomAddF(&deg[dst[i]], 1.0f);
}

__global__ void k_deg_rsqrt(float* __restrict__ deg, int n) {
  int i = blockIdx.x * 256 + threadIdx.x;
  if (i < n) deg[i] = rsqrtf(deg[i]);
}

// ---------------------------------------------------------------------------
// Pack f32 activations [N,256] into f16 A-operand layout for
// v_wmma_f32_16x16x32_f16. Per (mtile, kstep, lane): 16 halfs contiguous.
// ISA 7.12.2 (16-bit A 16x32): lane l -> M = l%16, half h = l/16;
//   elems 0..7  : K = kstep*32 + 8h + j
//   elems 8..15 : K = kstep*32 + 16 + 8h + j
// ---------------------------------------------------------------------------
__global__ void k_pack_A(const float* __restrict__ X, _Float16* __restrict__ XP, int nm) {
  int t = blockIdx.x * 256 + threadIdx.x;      // (mtile*8 + ks)*32 + lane
  if (t >= nm * 8 * 32) return;
  int lane  = t & 31;
  int ks    = (t >> 5) & 7;
  int mtile = t >> 8;
  int m     = mtile * 16 + (lane & 15);
  int kbase = ks * 32 + (lane >> 4) * 8;
  const float* row = X + (size_t)m * IN_FEATS;
  _Float16* out = XP + (size_t)t * 16;
#pragma unroll
  for (int j = 0; j < 8; ++j) out[j]     = (_Float16)row[kbase + j];
#pragma unroll
  for (int j = 0; j < 8; ++j) out[8 + j] = (_Float16)row[kbase + 16 + j];
}

// ---------------------------------------------------------------------------
// Pack f32 weights [256,F] row-major into f16 B-operand layout.
// B 32x16: lane l -> N = l%16; K = kstep*32 + 8*(l/16) + j (j<8), kstep*32 + 16 + 8*(l/16) + (j-8) (j>=8).
// ---------------------------------------------------------------------------
__global__ void k_pack_B(const float* __restrict__ W, _Float16* __restrict__ WP, int F) {
  int t = blockIdx.x * 256 + threadIdx.x;      // (ntile*8 + ks)*32 + lane
  if (t >= (F / 16) * 8 * 32) return;
  int lane  = t & 31;
  int ks    = (t >> 5) & 7;
  int ntile = t >> 8;
  int n     = ntile * 16 + (lane & 15);
  int kbase = ks * 32 + (lane >> 4) * 8;
  _Float16* out = WP + (size_t)t * 16;
#pragma unroll
  for (int j = 0; j < 16; ++j) out[j] = (_Float16)W[(size_t)(kbase + j + (j & 8)) * F + n];
}

// ---------------------------------------------------------------------------
// WMMA GEMM: H[N,F] = Xpacked @ Wpacked.  One wave per block.
// grid = (N/16, F/(16*NT)), block = 32.  Wave computes 16 x (16*NT) outputs,
// reusing one A fragment across NT B fragments. K = 256 -> 8 ksteps.
// All control flow wave-uniform -> EXEC stays all-ones for WMMA.
// ---------------------------------------------------------------------------
template <int NT>
__global__ void k_gemm_wmma(const _Float16* __restrict__ XP,
                            const _Float16* __restrict__ WP,
                            float* __restrict__ H, int F) {
  const int mtile = blockIdx.x;
  const int lane  = threadIdx.x;
  const int nt0   = blockIdx.y * NT;
  v8f acc[NT];
#pragma unroll
  for (int t = 0; t < NT; ++t) acc[t] = (v8f){0.f,0.f,0.f,0.f,0.f,0.f,0.f,0.f};

  const _Float16* aBase = XP + (size_t)mtile * 4096 + lane * 16;
#pragma unroll
  for (int ks = 0; ks < 8; ++ks) {
    v16h a = *(const v16h*)(aBase + ks * 512);
#pragma unroll
    for (int t = 0; t < NT; ++t) {
      v16h b = *(const v16h*)(WP + ((size_t)(nt0 + t) * 8 + ks) * 512 + lane * 16);
      acc[t] = __builtin_amdgcn_wmma_f32_16x16x32_f16(
          false, a, false, b, (short)0, acc[t], false, false);
    }
  }
  // C/D layout: lane l -> N=l%16; VGPR v -> M = v + 8*(l/16)
  const int half  = lane >> 4;
  const int nloc  = lane & 15;
  const int mbase = mtile * 16 + half * 8;
#pragma unroll
  for (int t = 0; t < NT; ++t) {
    const int n = (nt0 + t) * 16 + nloc;
#pragma unroll
    for (int v = 0; v < 8; ++v)
      H[(size_t)(mbase + v) * F + n] = acc[t][v];
  }
}

// ---------------------------------------------------------------------------
// Aggregation: self-loop init then edge scatter with native f32 atomics.
// ---------------------------------------------------------------------------
__global__ void k_agg_self(const float* __restrict__ H, float* __restrict__ AGG,
                           const float* __restrict__ dis, int n, int F) {
  int t = blockIdx.x * 256 + threadIdx.x;
  int total = n * (F >> 2);
  if (t >= total) return;
  int i = t / (F >> 2);
  float s = dis[i]; s = s * s;
  float4 h = ((const float4*)H)[t];
  float4 o = {h.x * s, h.y * s, h.z * s, h.w * s};
  ((float4*)AGG)[t] = o;
}

__global__ void k_agg_edges(const float* __restrict__ H, float* __restrict__ AGG,
                            const int* __restrict__ src, const int* __restrict__ dst,
                            const float* __restrict__ dis, int e, int F) {
  int chunks = F >> 2;
  long long t = (long long)blockIdx.x * 256 + threadIdx.x;
  if (t >= (long long)e * chunks) return;
  int ed = (int)(t / chunks);
  int c  = (int)(t % chunks);
  int s = src[ed], d = dst[ed];
  float nrm = dis[s] * dis[d];
  float4 h = *(const float4*)(H + (size_t)s * F + c * 4);
  float* out = AGG + (size_t)d * F + c * 4;
  atomAddF(out + 0, h.x * nrm);
  atomAddF(out + 1, h.y * nrm);
  atomAddF(out + 2, h.z * nrm);
  atomAddF(out + 3, h.w * nrm);
}

// HC = relu(AGG + b)                          (layer 1)
__global__ void k_bias_relu(const float* __restrict__ AGG, const float* __restrict__ b,
                            float* __restrict__ HC, int n) {
  int t = blockIdx.x * 256 + threadIdx.x;
  if (t >= n * H_FEATS) return;
  float v = AGG[t] + b[t & (H_FEATS - 1)];
  HC[t] = v > 0.f ? v : 0.f;
}

// HC = relu(relu(AGG + b) + HC)               (layers 2,3)
__global__ void k_resid_relu(const float* __restrict__ AGG, const float* __restrict__ b,
                             float* __restrict__ HC, int n) {
  int t = blockIdx.x * 256 + threadIdx.x;
  if (t >= n * H_FEATS) return;
  float v = AGG[t] + b[t & (H_FEATS - 1)];
  v = v > 0.f ? v : 0.f;
  v += HC[t];
  HC[t] = v > 0.f ? v : 0.f;
}

// ---------------------------------------------------------------------------
// Global mean pool over batch vector (64 graphs x 16 feats), b4 folded in.
// ---------------------------------------------------------------------------
__global__ void k_pool_zero(float* __restrict__ SUMS, float* __restrict__ CNT) {
  int t = blockIdx.x * 256 + threadIdx.x;
  if (t < NUM_GRAPHS * OUT_FEATS) SUMS[t] = 0.f;
  if (t < NUM_GRAPHS) CNT[t] = 0.f;
}

__global__ void k_pool_accum(const float* __restrict__ AGG16, const int* __restrict__ batch,
                             float* __restrict__ SUMS, float* __restrict__ CNT, int n) {
  int i = blockIdx.x * 256 + threadIdx.x;
  if (i >= n) return;
  int g = batch[i];
  const float* row = AGG16 + (size_t)i * OUT_FEATS;
  float* out = SUMS + (size_t)g * OUT_FEATS;
#pragma unroll
  for (int f = 0; f < OUT_FEATS; ++f) atomAddF(&out[f], row[f]);
  atomAddF(&CNT[g], 1.0f);
}

__global__ void k_pool_final(const float* __restrict__ SUMS, const float* __restrict__ CNT,
                             const float* __restrict__ b4, float* __restrict__ out) {
  int t = blockIdx.x * 256 + threadIdx.x;
  if (t >= NUM_GRAPHS * OUT_FEATS) return;
  int g = t / OUT_FEATS, f = t % OUT_FEATS;
  float c = CNT[g];
  out[t] = SUMS[t] / (c > 1.f ? c : 1.f) + b4[f];
}

// ---------------------------------------------------------------------------
extern "C" void kernel_launch(void* const* d_in, const int* in_sizes, int n_in,
                              void* d_out, int out_size, void* d_ws, size_t ws_size,
                              hipStream_t stream) {
  const float* x    = (const float*)d_in[0];
  const int*   src  = (const int*)d_in[1];
  const int*   batc = (const int*)d_in[2];
  const float* W1 = (const float*)d_in[3];  const float* b1 = (const float*)d_in[4];
  const float* W2 = (const float*)d_in[5];  const float* b2 = (const float*)d_in[6];
  const float* W3 = (const float*)d_in[7];  const float* b3 = (const float*)d_in[8];
  const float* W4 = (const float*)d_in[9];  const float* b4 = (const float*)d_in[10];
  float* out = (float*)d_out;

  const int N  = in_sizes[0] / IN_FEATS;   // 50000
  const int E  = in_sizes[1] / 2;          // 800000
  const int NM = N / 16;                   // 3125
  const int* dst = src + E;

  // workspace carve-up
  char* base = (char*)d_ws;
  size_t off = 0;
  auto carve = [&](size_t bytes) { char* p = base + off; off += (bytes + 255) & ~(size_t)255; return p; };
  float*     H    = (float*)carve((size_t)N * H_FEATS * 4);
  float*     AGG  = (float*)carve((size_t)N * H_FEATS * 4);
  float*     HC   = (float*)carve((size_t)N * H_FEATS * 4);
  _Float16*  XP   = (_Float16*)carve((size_t)N * H_FEATS * 2);
  _Float16*  W1p  = (_Float16*)carve(256 * 256 * 2);
  _Float16*  W2p  = (_Float16*)carve(256 * 256 * 2);
  _Float16*  W3p  = (_Float16*)carve(256 * 256 * 2);
  _Float16*  W4p  = (_Float16*)carve(256 * OUT_FEATS * 2);
  float*     dis  = (float*)carve((size_t)N * 4);
  float*     SUMS = (float*)carve(NUM_GRAPHS * OUT_FEATS * 4);
  float*     CNT  = (float*)carve(NUM_GRAPHS * 4);
  (void)ws_size; (void)n_in; (void)out_size;

  const int BS = 256;
  auto blocks = [](long long total) { return (int)((total + 255) / 256); };

  // --- pack weights into WMMA B layout (f16) ---
  k_pack_B<<<blocks(16 * 8 * 32), BS, 0, stream>>>(W1, W1p, 256);
  k_pack_B<<<blocks(16 * 8 * 32), BS, 0, stream>>>(W2, W2p, 256);
  k_pack_B<<<blocks(16 * 8 * 32), BS, 0, stream>>>(W3, W3p, 256);
  k_pack_B<<<blocks(1  * 8 * 32), BS, 0, stream>>>(W4, W4p, OUT_FEATS);

  // --- degrees ---
  k_deg_init <<<blocks(N), BS, 0, stream>>>(dis, N);
  k_deg_edges<<<blocks(E), BS, 0, stream>>>(dst, dis, E);
  k_deg_rsqrt<<<blocks(N), BS, 0, stream>>>(dis, N);

  const long long elemsH = (long long)N * H_FEATS;

  // --- layer 1 ---
  k_pack_A<<<blocks((long long)NM * 8 * 32), BS, 0, stream>>>(x, XP, NM);
  k_gemm_wmma<4><<<dim3(NM, 256 / 64), 32, 0, stream>>>(XP, W1p, H, 256);
  k_agg_self <<<blocks(elemsH / 4), BS, 0, stream>>>(H, AGG, dis, N, 256);
  k_agg_edges<<<blocks((long long)E * 64), BS, 0, stream>>>(H, AGG, src, dst, dis, E, 256);
  k_bias_relu<<<blocks(elemsH), BS, 0, stream>>>(AGG, b1, HC, N);

  // --- layer 2 ---
  k_pack_A<<<blocks((long long)NM * 8 * 32), BS, 0, stream>>>(HC, XP, NM);
  k_gemm_wmma<4><<<dim3(NM, 256 / 64), 32, 0, stream>>>(XP, W2p, H, 256);
  k_agg_self <<<blocks(elemsH / 4), BS, 0, stream>>>(H, AGG, dis, N, 256);
  k_agg_edges<<<blocks((long long)E * 64), BS, 0, stream>>>(H, AGG, src, dst, dis, E, 256);
  k_resid_relu<<<blocks(elemsH), BS, 0, stream>>>(AGG, b2, HC, N);

  // --- layer 3 ---
  k_pack_A<<<blocks((long long)NM * 8 * 32), BS, 0, stream>>>(HC, XP, NM);
  k_gemm_wmma<4><<<dim3(NM, 256 / 64), 32, 0, stream>>>(XP, W3p, H, 256);
  k_agg_self <<<blocks(elemsH / 4), BS, 0, stream>>>(H, AGG, dis, N, 256);
  k_agg_edges<<<blocks((long long)E * 64), BS, 0, stream>>>(H, AGG, src, dst, dis, E, 256);
  k_resid_relu<<<blocks(elemsH), BS, 0, stream>>>(AGG, b3, HC, N);

  // --- layer 4 (F=16): reuse H / AGG buffers at width 16 ---
  k_pack_A<<<blocks((long long)NM * 8 * 32), BS, 0, stream>>>(HC, XP, NM);
  k_gemm_wmma<1><<<dim3(NM, 1), 32, 0, stream>>>(XP, W4p, H, OUT_FEATS);
  k_agg_self <<<blocks((long long)N * OUT_FEATS / 4), BS, 0, stream>>>(H, AGG, dis, N, OUT_FEATS);
  k_agg_edges<<<blocks((long long)E * (OUT_FEATS / 4)), BS, 0, stream>>>(H, AGG, src, dst, dis, E, OUT_FEATS);

  // --- global mean pool (+ b4 folded into mean) ---
  k_pool_zero <<<blocks(NUM_GRAPHS * OUT_FEATS), BS, 0, stream>>>(SUMS, CNT);
  k_pool_accum<<<blocks(N), BS, 0, stream>>>(AGG, batc, SUMS, CNT, N);
  k_pool_final<<<blocks(NUM_GRAPHS * OUT_FEATS), BS, 0, stream>>>(SUMS, CNT, b4, out);
}
